// Memory_33706903339173
// MI455X (gfx1250) — matrix operand order repaired and checked by reading, not compiled, not run on hardware
//
#include <hip/hip_runtime.h>
#include <hip/hip_bf16.h>

// ---------------------------------------------------------------------------
// Memory-attention fused kernel for MI455X (gfx1250, wave32, WMMA)
//   x:       [32, 512, 64, 64] f32   (queries, channels strided by H*W=4096)
//   mempool: [100, 512] f32
//   out:     [32, 512, 64, 64] f32  ++  loss scalar (last element of d_out)
// Memory-bound: 536 MB streaming traffic -> ~23 us floor at 23.3 TB/s.
// ---------------------------------------------------------------------------

typedef __attribute__((ext_vector_type(16))) __bf16 bf16x16;
typedef __attribute__((ext_vector_type(8)))  __bf16 bf16x8;
typedef __attribute__((ext_vector_type(8)))  float  f32x8;

#define C_DIM   512
#define HW      4096          // H*W
#define M_ITEMS 100
#define M_PAD   112           // 7 * 16 column tiles
#define KTOP    10
#define ROWS    128           // pixels per workgroup (8 waves x 16 rows)
#define KCH     128           // K chunk streamed through LDS
#define LDQ     136           // bf16 elems per LDS row (136*2=272 B, 16B aligned)
#define LDA     104           // f32 att row stride (>=100)

// LDS layout (dynamic, 65280 B -> 4 concurrent WGs per 320KB WGP):
//  phase 1: qs[ROWS][LDQ] bf16 @ 0       (34816 B)
//           ms[M_PAD][LDQ] bf16 @ 34816  (30464 B)
//  phase 2 (reuses phase-1 space after GEMM):
//           att[ROWS][LDA] f32 @ 0       (53248 B)
//           widx[ROWS*KTOP] int @ 53248  (5120 B)
//           wval[ROWS*KTOP] f32 @ 58368  (5120 B)

// pack two f32 -> two bf16 in one dword (compiler may fuse to packed cvt)
static __device__ __forceinline__ unsigned pack2bf(float lo, float hi) {
    unsigned short a = __builtin_bit_cast(unsigned short, (__bf16)lo);
    unsigned short b = __builtin_bit_cast(unsigned short, (__bf16)hi);
    return ((unsigned)b << 16) | (unsigned)a;
}

__global__ __launch_bounds__(256) void mem_attn_kernel(
    const float* __restrict__ x, const float* __restrict__ mp,
    float* __restrict__ out)
{
    extern __shared__ char smem[];
    __bf16* qs  = (__bf16*)smem;                        // [ROWS][LDQ]
    __bf16* ms  = (__bf16*)(smem + ROWS * LDQ * 2);     // [M_PAD][LDQ]
    float*  att = (float*)smem;                         // [ROWS][LDA]
    int*    widx = (int*)(smem + ROWS * LDA * 4);
    float*  wval = (float*)(smem + ROWS * LDA * 4 + ROWS * KTOP * 4);

    const int t    = threadIdx.x;
    const int n0   = blockIdx.x * ROWS;      // first pixel of this tile
    const int b    = n0 >> 12;               // n0 / HW   (tile never crosses b)
    const int pix  = n0 & (HW - 1);
    const float* __restrict__ xbase = x + (size_t)b * (C_DIM * HW) + pix;

    const int lane = t & 31;
    const int wv   = t >> 5;                 // wave 0..7
    const int m0   = wv << 4;                // row base of this wave's stripe
    const int lhi  = lane >> 4;              // lane half (0/1)
    const int ln   = lane & 15;

    // 7 column tiles of 16x16 f32 accumulators per wave
    f32x8 acc[7];
    #pragma unroll
    for (int j = 0; j < 7; ++j)
        #pragma unroll
        for (int e = 0; e < 8; ++e) acc[j][e] = 0.0f;

    // ---------------- GEMM: logits = q . mempool^T (bf16 WMMA) -------------
    for (int kc = 0; kc < C_DIM; kc += KCH) {
        // stage q chunk: ROWS x KCH. Two coalesced NT b32 load streams
        // (adjacent channels), one packed 2xbf16 b32 DS store.
        for (int i = t; i < ROWS * (KCH / 2); i += 256) {
            int r  = i & (ROWS - 1);
            int cc = (i >> 7) << 1;
            float v0 = __builtin_nontemporal_load(&xbase[(size_t)(kc + cc) * HW + r]);
            float v1 = __builtin_nontemporal_load(&xbase[(size_t)(kc + cc + 1) * HW + r]);
            *(unsigned*)(qs + r * LDQ + cc) = pack2bf(v0, v1);
        }
        // stage mempool chunk: M_PAD x KCH. One b64 load, one packed b32 store.
        for (int i = t; i < M_PAD * (KCH / 2); i += 256) {
            int cc = (i & (KCH / 2 - 1)) << 1;
            int n  = i >> 6;                 // KCH/2 == 64
            float2 v = make_float2(0.0f, 0.0f);
            if (n < M_ITEMS) v = *(const float2*)(mp + n * C_DIM + kc + cc);
            *(unsigned*)(ms + n * LDQ + cc) = pack2bf(v.x, v.y);
        }
        __syncthreads();

        // speculative prefetch of the next q chunk (global_prefetch_b8)
        if (kc + KCH < C_DIM)
            __builtin_prefetch(xbase + (size_t)(kc + KCH + (t >> 1)) * HW
                                     + ((t & 1) << 6), 0, 3);

        #pragma unroll
        for (int ks = 0; ks < KCH; ks += 32) {
            // Preload A + B fragments, then issue the WMMA burst; scheduler
            // batches s_wait_dscnt as register pressure allows, and 8 waves
            // per SIMD cover the residual LDS latency.
            const __bf16* ap = qs + (m0 + ln) * LDQ + ks + (lhi << 3);
            bf16x16 A;
            *(bf16x8*)&A       = *(const bf16x8*)ap;         // K {0..7}+8*lhi
            *((bf16x8*)&A + 1) = *(const bf16x8*)(ap + 16);  // K {16..23}+8*lhi

            bf16x16 B[7];
            #pragma unroll
            for (int jt = 0; jt < 7; ++jt) {
                const __bf16* bp = ms + (jt * 16 + ln) * LDQ + ks + (lhi << 3);
                *(bf16x8*)&B[jt]       = *(const bf16x8*)bp;       // K {0..7}+8*lhi
                *((bf16x8*)&B[jt] + 1) = *(const bf16x8*)(bp + 16); // K {16..23}+8*lhi
            }

            #pragma unroll
            for (int jt = 0; jt < 7; ++jt)
                acc[jt] = __builtin_amdgcn_wmma_f32_16x16x32_bf16(
                    false, A, false, B[jt], (short)0, acc[jt], false, false);
        }
        __syncthreads();
    }

    // ------------- dump logits to LDS (reusing staging space) --------------
    #pragma unroll
    for (int jt = 0; jt < 7; ++jt)
        #pragma unroll
        for (int i = 0; i < 8; ++i) {
            int row = m0 + (lhi << 3) + i;      // C/D layout: VGPR i, lane half
            int col = jt * 16 + ln;
            att[row * LDA + col] = acc[jt][i];
        }
    __syncthreads();

    // -------- per-row softmax -> top-10 -> renormalizing softmax -----------
    if (t < ROWS) {
        float* rowp = att + t * LDA;
        float mx = -3.4e38f;
        for (int j = 0; j < M_ITEMS; ++j) mx = fmaxf(mx, rowp[j]);
        float s = 0.0f;
        for (int j = 0; j < M_ITEMS; ++j) {
            float e = __expf(rowp[j] - mx);
            rowp[j] = e; s += e;
        }
        float inv = 1.0f / s;

        float v[KTOP]; int id[KTOP];
        #pragma unroll
        for (int k = 0; k < KTOP; ++k) {
            float bm = -1.0f; int bi = 0;
            for (int j = 0; j < M_ITEMS; ++j) {
                float e = rowp[j];
                if (e > bm) { bm = e; bi = j; }
            }
            rowp[bi] = -1.0f;                 // remove selected
            v[k] = bm * inv; id[k] = bi;      // probability + index
        }
        // second softmax over the top-k probabilities (v[0] is the max)
        float s2 = 0.0f;
        #pragma unroll
        for (int k = 0; k < KTOP; ++k) { float e = __expf(v[k] - v[0]); v[k] = e; s2 += e; }
        float i2 = 1.0f / s2;
        #pragma unroll
        for (int k = 0; k < KTOP; ++k) {
            widx[t * KTOP + k] = id[k];
            wval[t * KTOP + k] = v[k] * i2;
        }
    }
    __syncthreads();

    // ------------- reconstruct: out[n][c] = sum_k w_k * mempool[idx_k][c] ---
    {
        const int rl   = t & (ROWS - 1);
        const int half = t >> 7;              // 2 threads per row, 256 ch each
        int   id[KTOP]; float w[KTOP];
        #pragma unroll
        for (int k = 0; k < KTOP; ++k) {
            id[k] = widx[rl * KTOP + k];
            w[k]  = wval[rl * KTOP + k];
        }
        float* obase = out + (size_t)b * (C_DIM * HW) + pix + rl;
        const int c0 = half * 256;
        for (int c = c0; c < c0 + 256; c += 4) {
            float ax = 0.f, ay = 0.f, az = 0.f, aw = 0.f;
            #pragma unroll
            for (int k = 0; k < KTOP; ++k) {
                const float4 m4 = *(const float4*)(mp + id[k] * C_DIM + c);
                ax += w[k] * m4.x; ay += w[k] * m4.y;
                az += w[k] * m4.z; aw += w[k] * m4.w;
            }
            // write-once stream: non-temporal stores, coalesced across lanes
            __builtin_nontemporal_store(ax, &obase[(size_t)(c + 0) * HW]);
            __builtin_nontemporal_store(ay, &obase[(size_t)(c + 1) * HW]);
            __builtin_nontemporal_store(az, &obase[(size_t)(c + 2) * HW]);
            __builtin_nontemporal_store(aw, &obase[(size_t)(c + 3) * HW]);
        }
    }
}

// ------------------- diversity loss: mean |0.5 * off-diag gram| ------------
__global__ __launch_bounds__(256) void mem_loss_kernel(
    const float* __restrict__ mp, float* __restrict__ loss_out)
{
    __shared__ float red[256];
    const int t = threadIdx.x;
    float acc = 0.0f;
    for (int p = t; p < M_ITEMS * M_ITEMS; p += 256) {
        int i = p / M_ITEMS, j = p % M_ITEMS;
        if (i != j) {
            const float* a  = mp + i * C_DIM;
            const float* bb = mp + j * C_DIM;
            float d = 0.0f;
            for (int c = 0; c < C_DIM; c += 4) {
                float4 av = *(const float4*)(a + c);
                float4 bv = *(const float4*)(bb + c);
                d += av.x * bv.x + av.y * bv.y + av.z * bv.z + av.w * bv.w;
            }
            acc += fabsf(0.5f * d);
        }
    }
    red[t] = acc;
    __syncthreads();
    for (int s = 128; s > 0; s >>= 1) {
        if (t < s) red[t] += red[t + s];
        __syncthreads();
    }
    if (t == 0) *loss_out = red[0] / (float)(M_ITEMS * M_ITEMS);
}

extern "C" void kernel_launch(void* const* d_in, const int* in_sizes, int n_in,
                              void* d_out, int out_size, void* d_ws, size_t ws_size,
                              hipStream_t stream) {
    (void)n_in; (void)d_ws; (void)ws_size;
    const float* x  = (const float*)d_in[0];   // [32,512,64,64] f32
    const float* mp = (const float*)d_in[1];   // [100,512] f32
    float* out = (float*)d_out;

    const int N    = in_sizes[0] / C_DIM;      // 131072 pixels
    const int grid = N / ROWS;                 // 1024 workgroups
    const size_t ldsBytes = 65280;             // max(phase1 65280, phase2 63488)

    mem_attn_kernel<<<grid, 256, ldsBytes, stream>>>(x, mp, out);
    mem_loss_kernel<<<1, 256, 0, stream>>>(mp, out + (size_t)out_size - 1);
}
